// SelfAttention_457
// MI455X (gfx1250) — matrix operand order repaired and checked by reading, not compiled, not run on hardware
//
#include <hip/hip_runtime.h>
#include <stdint.h>

#define N_BATCH 2
#define SEQ 2048
#define EMBED 1024
#define HEADS 16
#define HEAD_DIM 64
#define SOFT_SCALE 0.03125f   // 1/sqrt(EMBED) = 1/32

typedef __attribute__((ext_vector_type(16))) _Float16 v16h;
typedef __attribute__((ext_vector_type(8)))  _Float16 v8h;
typedef __attribute__((ext_vector_type(8)))  float    v8f;
typedef __attribute__((ext_vector_type(4)))  float    v4f;

static __device__ __forceinline__ v8f wmma16(v16h a, v16h b, v8f c) {
  // D = A(16x32 f16) * B(32x16 f16) + C(16x16 f32)
  return __builtin_amdgcn_wmma_f32_16x16x32_f16(false, a, false, b, (short)0, c,
                                                false, false);
}

static __device__ __forceinline__ v16h cat16(v8h lo, v8h hi) {
  return __builtin_shufflevector(lo, hi, 0, 1, 2, 3, 4, 5, 6, 7,
                                 8, 9, 10, 11, 12, 13, 14, 15);
}

static __device__ __forceinline__ v8h cvt8_f32_to_h(const float* p) {
  v4f x0 = *(const v4f*)p;
  v4f x1 = *(const v4f*)(p + 4);
  v8h r;
#pragma unroll
  for (int i = 0; i < 4; ++i) {
    r[i]     = (_Float16)x0[i];
    r[i + 4] = (_Float16)x1[i];
  }
  return r;
}

// gfx1250 async copy: LDS[lds] = MEM[gaddr], 16 bytes per lane, ASYNCcnt.
static __device__ __forceinline__ void async_cp16(unsigned lds, uint64_t gaddr) {
  asm volatile("global_load_async_to_lds_b128 %0, %1, off"
               :: "v"(lds), "v"(gaddr)
               : "memory");
}
static __device__ __forceinline__ void wait_async_le16() {
  asm volatile("s_wait_asynccnt 0x10" ::: "memory");
}
static __device__ __forceinline__ void wait_async_0() {
  asm volatile("s_wait_asynccnt 0x0" ::: "memory");
}

// ---------------------------------------------------------------------------
// fp32 -> f16 elementwise convert (for Wo)
// ---------------------------------------------------------------------------
__global__ void cvt_f32_to_f16_kernel(const float* __restrict__ src,
                                      _Float16* __restrict__ dst, int n) {
  int i = blockIdx.x * blockDim.x + threadIdx.x;
  if (i < n) dst[i] = (_Float16)src[i];
}

// ---------------------------------------------------------------------------
// Per-head input projection: Y[r][e] = sum_d X[r][h*64+d] * W[e][d]
// One wave: 16 rows x all 64 output dims for one head. 8 WMMAs.
// transposeStore=0 -> dst[(n*H+h)*S + s][e]   (Q, K layout)
// transposeStore=1 -> dst[(n*H+h)*64 + e][s]  (V transposed layout)
// ---------------------------------------------------------------------------
__global__ __launch_bounds__(32) void proj_kernel(const float* __restrict__ X,
                                                  const float* __restrict__ W,
                                                  _Float16* __restrict__ dst,
                                                  int transposeStore) {
  const int lane = threadIdx.x;
  const int half = lane >> 4;
  const int l16  = lane & 15;
  const int rowBase = blockIdx.x * 16;
  const int h = blockIdx.y;

  // A operand (rows of X, f32 -> f16), contiguous k: half h holds k = 16h..16h+15
  v16h a[2];
#pragma unroll
  for (int c = 0; c < 2; ++c) {
    const float* p =
        X + (size_t)(rowBase + l16) * EMBED + h * HEAD_DIM + c * 32 + half * 16;
    a[c] = cat16(cvt8_f32_to_h(p), cvt8_f32_to_h(p + 8));
  }

  v8f acc[4] = {};
#pragma unroll
  for (int t = 0; t < 4; ++t) {
#pragma unroll
    for (int c = 0; c < 2; ++c) {
      // B[k=d][n=e] = W[e][d]; lane holds column e = t*16+l16, contiguous d
      const float* wp =
          W + (size_t)(t * 16 + l16) * HEAD_DIM + c * 32 + half * 16;
      v16h b = cat16(cvt8_f32_to_h(wp), cvt8_f32_to_h(wp + 8));
      acc[t] = wmma16(a[c], b, acc[t]);
    }
  }

#pragma unroll
  for (int t = 0; t < 4; ++t) {
#pragma unroll
    for (int i = 0; i < 8; ++i) {
      int row = rowBase + half * 8 + i;   // C-layout: M = i (+8 for hi half)
      int n = row / SEQ, s = row % SEQ;
      int e = t * 16 + l16;
      size_t idx;
      if (transposeStore)
        idx = ((size_t)(n * HEADS + h) * HEAD_DIM + e) * SEQ + s;
      else
        idx = ((size_t)(n * HEADS + h) * SEQ + s) * HEAD_DIM + e;
      dst[idx] = (_Float16)acc[t][i];
    }
  }
}

// ---------------------------------------------------------------------------
// Flash attention: one wave = 16 queries of one (n,h). Online softmax over
// 32-key chunks. K/V chunks are double-buffered in LDS via async-to-LDS
// copies (ASYNCcnt); P is converted C-layout -> A-layout via a 1KB LDS tile.
// ---------------------------------------------------------------------------
__global__ __launch_bounds__(32) void attn_kernel(
    const _Float16* __restrict__ Qp, const _Float16* __restrict__ Kp,
    const _Float16* __restrict__ Vt, _Float16* __restrict__ AO) {
  const int lane = threadIdx.x;
  const int half = lane >> 4;
  const int l16  = lane & 15;
  const int qBase = blockIdx.x * 16;
  const int nh = blockIdx.y;
  const int n = nh / HEADS, h = nh % HEADS;

  const _Float16* Qh = Qp + (size_t)nh * SEQ * HEAD_DIM;
  const _Float16* Kh = Kp + (size_t)nh * SEQ * HEAD_DIM;
  const _Float16* Vh = Vt + (size_t)nh * HEAD_DIM * SEQ;

  __shared__ __align__(16) _Float16 Kl[2][32][64];  // [buf][key][d]     4KB/buf
  __shared__ __align__(16) _Float16 Vl[2][64][32];  // [buf][dout][key]  4KB/buf
  __shared__ __align__(16) _Float16 Pl[16][32];

  // Per-lane async-copy bases (byte addresses). K chunk = contiguous 4KB.
  const unsigned ldsK = (unsigned)(uintptr_t)&Kl[0][0][0] + (unsigned)lane * 16u;
  const unsigned ldsV = (unsigned)(uintptr_t)&Vl[0][0][0] +
                        (unsigned)(lane >> 2) * 64u + (unsigned)(lane & 3) * 16u;
  uint64_t gK = (uint64_t)(uintptr_t)Kh + (uint64_t)lane * 16u;
  uint64_t gV = (uint64_t)(uintptr_t)Vh + (uint64_t)(lane >> 2) * (SEQ * 2) +
                (uint64_t)(lane & 3) * 16u;

  // ---- stage chunk 0 into buffer 0 ----
#pragma unroll
  for (int j = 0; j < 8; ++j)
    async_cp16(ldsK + j * 512u, gK + (uint64_t)j * 512u);
#pragma unroll
  for (int j = 0; j < 8; ++j)
    async_cp16(ldsV + j * 512u, gV + (uint64_t)j * (8u * SEQ * 2));
  gK += 32 * HEAD_DIM * 2;  // next 32 keys
  gV += 32 * 2;

  // Q tile A-operands (reused for all key chunks)
  v16h aq[2];
#pragma unroll
  for (int c = 0; c < 2; ++c) {
    const _Float16* p =
        Qh + (size_t)(qBase + l16) * HEAD_DIM + c * 32 + half * 16;
    aq[c] = cat16(*(const v8h*)p, *(const v8h*)(p + 8));
  }

  v8f acc[4] = {};
  float rowm[8], rowl[8];
#pragma unroll
  for (int i = 0; i < 8; ++i) { rowm[i] = -3.0e38f; rowl[i] = 0.0f; }

  const int NC = SEQ / 32;
  for (int kc = 0; kc < NC; ++kc) {
    const int buf = kc & 1;

    if (kc + 1 < NC) {
      // stage next chunk into the other buffer while this one computes
      const unsigned bo = (unsigned)(buf ^ 1) * 4096u;
#pragma unroll
      for (int j = 0; j < 8; ++j)
        async_cp16(ldsK + bo + j * 512u, gK + (uint64_t)j * 512u);
#pragma unroll
      for (int j = 0; j < 8; ++j)
        async_cp16(ldsV + bo + j * 512u, gV + (uint64_t)j * (8u * SEQ * 2));
      gK += 32 * HEAD_DIM * 2;
      gV += 32 * 2;
      wait_async_le16();   // chunk kc resident (async loads complete in order)
    } else {
      wait_async_0();
    }

    // ---- energy: two 16x16 tiles of Q @ K^T over D=64, K from LDS ----
    v8f e0 = {}, e1 = {};
#pragma unroll
    for (int c = 0; c < 2; ++c) {
      const _Float16* k0 = &Kl[buf][l16][c * 32 + half * 16];
      const _Float16* k1 = &Kl[buf][16 + l16][c * 32 + half * 16];
      e0 = wmma16(aq[c], cat16(*(const v8h*)k0, *(const v8h*)(k0 + 8)), e0);
      e1 = wmma16(aq[c], cat16(*(const v8h*)k1, *(const v8h*)(k1 + 8)), e1);
    }

    // ---- online softmax update (row = half*8 + i, cols across lanes) ----
    float p0[8], p1[8], fr[8];
#pragma unroll
    for (int i = 0; i < 8; ++i) {
      float s0 = e0[i] * SOFT_SCALE;
      float s1 = e1[i] * SOFT_SCALE;
      float mx = fmaxf(s0, s1);
#pragma unroll
      for (int off = 8; off > 0; off >>= 1)  // stays within 16-lane half
        mx = fmaxf(mx, __shfl_xor(mx, off, 32));
      float mnew = fmaxf(rowm[i], mx);
      float v0 = __expf(s0 - mnew);
      float v1 = __expf(s1 - mnew);
      float sum = v0 + v1;
#pragma unroll
      for (int off = 8; off > 0; off >>= 1)
        sum += __shfl_xor(sum, off, 32);
      float f = __expf(rowm[i] - mnew);
      rowl[i] = rowl[i] * f + sum;
      rowm[i] = mnew;
      fr[i] = f;
      p0[i] = v0;
      p1[i] = v1;
    }
#pragma unroll
    for (int t = 0; t < 4; ++t)
#pragma unroll
      for (int i = 0; i < 8; ++i) acc[t][i] *= fr[i];

    // ---- C-layout P -> LDS [query row][key-in-chunk] ----
#pragma unroll
    for (int i = 0; i < 8; ++i) {
      int r = half * 8 + i;
      Pl[r][l16]      = (_Float16)p0[i];
      Pl[r][16 + l16] = (_Float16)p1[i];
    }
    __syncthreads();

    // ---- reload P as A-operand (16x32), then P @ V into accumulators ----
    const _Float16* pp = &Pl[l16][half * 16];
    v16h ap = cat16(*(const v8h*)pp, *(const v8h*)(pp + 8));
#pragma unroll
    for (int t = 0; t < 4; ++t) {
      // B[k=key][n=dout] = Vl[dout][key]; lane holds column dout = t*16+l16
      const _Float16* vp = &Vl[buf][t * 16 + l16][half * 16];
      acc[t] = wmma16(ap, cat16(*(const v8h*)vp, *(const v8h*)(vp + 8)), acc[t]);
    }
    __syncthreads();
  }

  // ---- normalize and store f16 [n][q][h*64+e] ----
#pragma unroll
  for (int t = 0; t < 4; ++t) {
#pragma unroll
    for (int i = 0; i < 8; ++i) {
      int q = qBase + half * 8 + i;
      int e = h * HEAD_DIM + t * 16 + l16;
      AO[((size_t)n * SEQ + q) * EMBED + e] = (_Float16)(acc[t][i] / rowl[i]);
    }
  }
}

// ---------------------------------------------------------------------------
// Output projection: out[r][e] = sum_d AO[r][d] * Wo[e][d] + bo[e]
// One wave = one 16x16 tile, K = 1024 (32 WMMAs).
// ---------------------------------------------------------------------------
__global__ __launch_bounds__(32) void outproj_kernel(
    const _Float16* __restrict__ AO, const _Float16* __restrict__ Woh,
    const float* __restrict__ bo, float* __restrict__ out) {
  const int lane = threadIdx.x;
  const int half = lane >> 4;
  const int l16  = lane & 15;
  const int rowBase = blockIdx.x * 16;
  const int eBase   = blockIdx.y * 16;

  v8f acc = {};
#pragma unroll 4
  for (int c = 0; c < EMBED / 32; ++c) {
    const _Float16* pa =
        AO + (size_t)(rowBase + l16) * EMBED + c * 32 + half * 16;
    v16h a = cat16(*(const v8h*)pa, *(const v8h*)(pa + 8));
    const _Float16* pb =
        Woh + (size_t)(eBase + l16) * EMBED + c * 32 + half * 16;
    v16h b = cat16(*(const v8h*)pb, *(const v8h*)(pb + 8));
    acc = wmma16(a, b, acc);
  }

  float bias = bo[eBase + l16];
#pragma unroll
  for (int i = 0; i < 8; ++i) {
    int row = rowBase + half * 8 + i;
    out[(size_t)row * EMBED + eBase + l16] = acc[i] + bias;
  }
}

// ---------------------------------------------------------------------------
extern "C" void kernel_launch(void* const* d_in, const int* in_sizes, int n_in,
                              void* d_out, int out_size, void* d_ws,
                              size_t ws_size, hipStream_t stream) {
  (void)in_sizes; (void)n_in; (void)out_size; (void)ws_size;
  const float* values = (const float*)d_in[0];
  const float* keys   = (const float*)d_in[1];
  const float* query  = (const float*)d_in[2];
  const float* Wv     = (const float*)d_in[3];
  const float* Wk     = (const float*)d_in[4];
  const float* Wq     = (const float*)d_in[5];
  const float* Wo     = (const float*)d_in[6];
  const float* bo     = (const float*)d_in[7];
  float* out = (float*)d_out;

  const size_t NHSD = (size_t)N_BATCH * HEADS * SEQ * HEAD_DIM;  // 4M halves
  _Float16* Qp  = (_Float16*)d_ws;
  _Float16* Kp  = Qp + NHSD;
  _Float16* Vt  = Kp + NHSD;
  _Float16* AO  = Vt + NHSD;
  _Float16* Woh = AO + (size_t)N_BATCH * SEQ * EMBED;

  cvt_f32_to_f16_kernel<<<(EMBED * EMBED) / 256, 256, 0, stream>>>(
      Wo, Woh, EMBED * EMBED);

  dim3 pgrid((N_BATCH * SEQ) / 16, HEADS);
  proj_kernel<<<pgrid, 32, 0, stream>>>(query,  Wq, Qp, 0);
  proj_kernel<<<pgrid, 32, 0, stream>>>(keys,   Wk, Kp, 0);
  proj_kernel<<<pgrid, 32, 0, stream>>>(values, Wv, Vt, 1);

  attn_kernel<<<dim3(SEQ / 16, N_BATCH * HEADS), 32, 0, stream>>>(Qp, Kp, Vt,
                                                                  AO);

  outproj_kernel<<<dim3((N_BATCH * SEQ) / 16, EMBED / 16), 32, 0, stream>>>(
      AO, Woh, bo, out);
}
